// VolatilityModulatedAttention_29549374996880
// MI455X (gfx1250) — matrix operand order repaired and checked by reading, not compiled, not run on hardware
//
#include <hip/hip_runtime.h>
#include <hip/hip_bf16.h>
#include <math.h>

// ---------------------------------------------------------------------------
// Problem constants (from reference): B=2, S=4096, H=512, NUM_HEADS=8, hd=64
// ---------------------------------------------------------------------------
#define BB 2
#define SS 4096
#define HH 512
#define NH 8
#define HD 64

#define INVLN2 1.4426950408889634f

typedef _Float16 half_t;
typedef __attribute__((ext_vector_type(16))) _Float16 v16h;
typedef __attribute__((ext_vector_type(8)))  _Float16 v8h;
typedef __attribute__((ext_vector_type(8)))  float    v8f;

static __device__ __forceinline__ v8f wmma_f16(v16h a, v16h b, v8f c) {
    // D = A(16x32 f16) * B(32x16 f16) + C(16x16 f32)
    return __builtin_amdgcn_wmma_f32_16x16x32_f16(
        /*neg_a=*/false, a, /*neg_b=*/false, b,
        /*c_mod=*/(short)0, c, /*reuse_a=*/false, /*reuse_b=*/false);
}

static __device__ __forceinline__ v16h combine16(v8h lo, v8h hi) {
    v16h r;
#pragma unroll
    for (int i = 0; i < 8; ++i) { r[i] = lo[i]; r[i + 8] = hi[i]; }
    return r;
}

// ---------------------------------------------------------------------------
// DPP-based xor-shuffle reductions within 16-lane groups (no LDS traffic).
//   xor1 -> quad_perm(1,0,3,2)=0xB1, xor2 -> quad_perm(2,3,0,1)=0x4E,
//   xor4 -> ROW_HALF_MIRROR=0x141,   xor8 -> ROW_MIRROR=0x140
// ---------------------------------------------------------------------------
template <int CTRL>
static __device__ __forceinline__ float dppf(float x) {
    return __builtin_bit_cast(float,
        __builtin_amdgcn_mov_dpp(__builtin_bit_cast(int, x), CTRL, 0xF, 0xF, true));
}
static __device__ __forceinline__ float red_max16(float x) {
    x = fmaxf(x, dppf<0xB1>(x));
    x = fmaxf(x, dppf<0x4E>(x));
    x = fmaxf(x, dppf<0x141>(x));
    x = fmaxf(x, dppf<0x140>(x));
    return x;
}
static __device__ __forceinline__ float red_sum16(float x) {
    x += dppf<0xB1>(x);
    x += dppf<0x4E>(x);
    x += dppf<0x141>(x);
    x += dppf<0x140>(x);
    return x;
}

// ---------------------------------------------------------------------------
// Kernel 1: pack f32 -> f16, head-major Q/K, transposed V, f16 W_out.
//   Qh     : [B, NH, S, HD], pre-scaled by (1/sqrt(hd)) * (1/ln2) so that
//            QK^T scores come out of WMMA already in the exp2 domain.
//   Kh     : [B, NH, S, HD]
//   Vt     : [B, NH, HD, S]   (dim-major so PV B-fragments are contiguous)
// ---------------------------------------------------------------------------
__global__ void __launch_bounds__(256)
pack_qkv_kernel(const float* __restrict__ Q, const float* __restrict__ K,
                const float* __restrict__ V,
                half_t* __restrict__ Qh, half_t* __restrict__ Kh,
                half_t* __restrict__ Vt)
{
    size_t i = (size_t)blockIdx.x * blockDim.x + threadIdx.x;
    const size_t total = (size_t)BB * SS * HH;
    if (i >= total) return;
    int hc = (int)(i % HH);
    size_t bs = i / HH;
    int s = (int)(bs % SS);
    int b = (int)(bs / SS);
    int head = hc / HD, d = hc % HD;
    size_t qk = (((size_t)b * NH + head) * SS + s) * HD + d;
    const float qscale = 0.125f * INVLN2;     // 1/sqrt(64) * 1/ln2
    Qh[qk] = (half_t)(Q[i] * qscale);
    Kh[qk] = (half_t)K[i];
    Vt[(((size_t)b * NH + head) * HD + d) * SS + s] = (half_t)V[i];
}

__global__ void __launch_bounds__(256)
pack_w_kernel(const float* __restrict__ W, half_t* __restrict__ Wh)
{
    int i = blockIdx.x * blockDim.x + threadIdx.x;
    if (i < HH * HH) Wh[i] = (half_t)W[i];
}

// ---------------------------------------------------------------------------
// Kernel 2: flash attention with distance-risk bias (log2-domain softmax).
// Grid: B*NH*(S/64) blocks, 128 threads (4 waves); each wave owns 16 rows.
// ---------------------------------------------------------------------------
__global__ void __launch_bounds__(128)
flash_attn_kernel(const half_t* __restrict__ Qh, const half_t* __restrict__ Kh,
                  const half_t* __restrict__ Vt,
                  const float* __restrict__ risk_sigma,
                  const float* __restrict__ risk_lambda,
                  half_t* __restrict__ Ctx)
{
    __shared__ half_t pshuf[4][16 * 32];   // per-wave P staging (C->A layout)

    const int lane   = threadIdx.x & 31;
    const int wave   = threadIdx.x >> 5;
    const int l16    = lane & 15;
    const int half16 = lane >> 4;          // 0: lanes 0-15, 1: lanes 16-31
    const int gA = half16 * 8;             // unused; A fragments also load at gB
    const int gB = half16 * 16;            // B-fragment K-offset

    const int tiles_per_seq = SS / 64;
    int blk  = blockIdx.x;
    int tile = blk % tiles_per_seq;
    int bh   = blk / tiles_per_seq;
    int head = bh % NH;
    int b    = bh / NH;
    const int row0 = tile * 64 + wave * 16;     // first query row of this wave

    const half_t* qbase = Qh + ((size_t)(b * NH + head) * SS) * HD;
    const half_t* kbase = Kh + ((size_t)(b * NH + head) * SS) * HD;
    const half_t* vbase = Vt + ((size_t)(b * NH + head) * HD) * SS;

    // risk bias coefficient: lam * sigma_n * (|i-j|/S)^2, folded into log2 domain
    float sg = risk_sigma[b];
    sg = fminf(fmaxf(sg, 0.001f), 0.2f);
    float sn   = (sg - 0.001f) * (1.0f / (0.2f - 0.001f));
    float coef = risk_lambda[0] * 0.1f * sn * INVLN2 / ((float)SS * (float)SS);

    // Q A-fragments (16x32 f16 per chunk), dims [0,32) and [32,64)
    const half_t* qrow = qbase + (size_t)(row0 + l16) * HD;
    v16h qa0 = *(const v16h*)(qrow + gB);
    v16h qa1 = *(const v16h*)(qrow + 32 + gB);

    v8f o0 = {}, o1 = {}, o2 = {}, o3 = {};
    float m[8], lsum[8], fd[8];
#pragma unroll
    for (int r = 0; r < 8; ++r) {
        m[r]    = -INFINITY;
        lsum[r] = 0.0f;
        // running signed distance row - (j + l16); decreases by 32 per j-step
        fd[r]   = (float)(row0 + r + 8 * half16 - l16);
    }

    half_t* pw = &pshuf[wave][0];

    for (int j = 0; j < SS; j += 32) {
        // ---- S = Q * K^T for 32 keys (two 16-key N-tiles) -----------------
        const half_t* kr0 = kbase + (size_t)(j + l16) * HD;        // keys j..j+15
        const half_t* kr1 = kbase + (size_t)(j + 16 + l16) * HD;   // keys j+16..j+31
        v16h kb00 = *(const v16h*)(kr0 + gB);        // dims [0,32)
        v16h kb01 = *(const v16h*)(kr0 + 32 + gB);   // dims [32,64)
        v16h kb10 = *(const v16h*)(kr1 + gB);
        v16h kb11 = *(const v16h*)(kr1 + 32 + gB);
        if (j + 32 < SS)  // hint next K tile toward the caches
            __builtin_prefetch(kbase + (size_t)(j + 32 + l16) * HD, 0, 3);

        v8f s0 = {}, s1 = {};
        s0 = wmma_f16(qa0, kb00, s0);
        s0 = wmma_f16(qa1, kb01, s0);
        s1 = wmma_f16(qa0, kb10, s1);
        s1 = wmma_f16(qa1, kb11, s1);

        // ---- online softmax (log2 domain) with risk bias ------------------
#pragma unroll
        for (int r = 0; r < 8; ++r) {
            float d0 = fd[r];
            float d1 = fd[r] - 16.0f;
            float v0 = fmaf(-coef * d0, d0, s0[r]);   // scores already *1/ln2
            float v1 = fmaf(-coef * d1, d1, s1[r]);
            float mx = red_max16(fmaxf(v0, v1));
            float mnew = fmaxf(m[r], mx);
            float p0 = __builtin_amdgcn_exp2f(v0 - mnew);
            float p1 = __builtin_amdgcn_exp2f(v1 - mnew);
            float ps = red_sum16(p0 + p1);
            float alpha = __builtin_amdgcn_exp2f(m[r] - mnew);
            lsum[r] = lsum[r] * alpha + ps;
            m[r] = mnew;
            fd[r] = d0 - 32.0f;
            o0[r] *= alpha; o1[r] *= alpha; o2[r] *= alpha; o3[r] *= alpha;
            // stage P (C-layout element -> row-major LDS)
            int prow = r + 8 * half16;
            pw[prow * 32 + l16]      = (half_t)p0;
            pw[prow * 32 + 16 + l16] = (half_t)p1;
        }
        // same-wave LDS RAW: drain DS counter before re-reading
        asm volatile("s_wait_dscnt 0" ::: "memory");

        // ---- reload P as A-fragment (16x32), accumulate O += P * V --------
        v16h pa = *(const v16h*)(pw + l16 * 32 + gB);
        v16h vf0 = *(const v16h*)(vbase + (size_t)(0 * 16 + l16) * SS + j + gB);
        v16h vf1 = *(const v16h*)(vbase + (size_t)(1 * 16 + l16) * SS + j + gB);
        v16h vf2 = *(const v16h*)(vbase + (size_t)(2 * 16 + l16) * SS + j + gB);
        v16h vf3 = *(const v16h*)(vbase + (size_t)(3 * 16 + l16) * SS + j + gB);
        o0 = wmma_f16(pa, vf0, o0);
        o1 = wmma_f16(pa, vf1, o1);
        o2 = wmma_f16(pa, vf2, o2);
        o3 = wmma_f16(pa, vf3, o3);
    }

    // ---- normalize and store ctx as f16 [B, S, H] -------------------------
#pragma unroll
    for (int r = 0; r < 8; ++r) {
        float inv = 1.0f / lsum[r];
        int row = row0 + r + 8 * half16;
        size_t base = ((size_t)b * SS + row) * HH + head * HD + l16;
        Ctx[base + 0]  = (half_t)(o0[r] * inv);
        Ctx[base + 16] = (half_t)(o1[r] * inv);
        Ctx[base + 32] = (half_t)(o2[r] * inv);
        Ctx[base + 48] = (half_t)(o3[r] * inv);
    }
}

// ---------------------------------------------------------------------------
// Kernel 3: output projection  out = ctx @ W_out^T + b_out  (f16 WMMA, f32 out)
// One wave per 16x16 output tile; 256-thread blocks (8 waves).
// ---------------------------------------------------------------------------
__global__ void __launch_bounds__(256)
proj_kernel(const half_t* __restrict__ Ctx, const half_t* __restrict__ Wh,
            const float* __restrict__ b_out, float* __restrict__ Out)
{
    const int lane   = threadIdx.x & 31;
    const int wave   = threadIdx.x >> 5;
    const int l16    = lane & 15;
    const int half16 = lane >> 4;
    const int gA = half16 * 8;
    const int gB = half16 * 16;

    const int htiles = HH / 16;                         // 32
    int tile   = blockIdx.x * 8 + wave;                 // (B*S/16) * htiles tiles
    int tile_h = tile % htiles;
    int tile_s = tile / htiles;                         // flat over B*S rows / 16

    const half_t* arow = Ctx + (size_t)(tile_s * 16 + l16) * HH;
    const half_t* brow = Wh  + (size_t)(tile_h * 16 + l16) * HH;

    v8f acc = {};
#pragma unroll
    for (int kc = 0; kc < HH; kc += 32) {
        v16h a = *(const v16h*)(arow + kc + gB);
        v16h bf = *(const v16h*)(brow + kc + gB);
        acc = wmma_f16(a, bf, acc);
    }

    float bo = b_out[tile_h * 16 + l16];
#pragma unroll
    for (int r = 0; r < 8; ++r) {
        size_t row = (size_t)(tile_s * 16 + r + 8 * half16);
        Out[row * HH + tile_h * 16 + l16] = acc[r] + bo;
    }
}

// ---------------------------------------------------------------------------
// Host-side launch
// inputs: 0=Q 1=K 2=V 3=risk_sigma 4=risk_lambda 5=W_out 6=b_out
// ---------------------------------------------------------------------------
extern "C" void kernel_launch(void* const* d_in, const int* in_sizes, int n_in,
                              void* d_out, int out_size, void* d_ws, size_t ws_size,
                              hipStream_t stream) {
    const float* Q  = (const float*)d_in[0];
    const float* K  = (const float*)d_in[1];
    const float* V  = (const float*)d_in[2];
    const float* sg = (const float*)d_in[3];
    const float* lm = (const float*)d_in[4];
    const float* W  = (const float*)d_in[5];
    const float* bo = (const float*)d_in[6];
    float* Out = (float*)d_out;

    const size_t QK = (size_t)BB * SS * HH;   // 4,194,304 elements
    half_t* Qh  = (half_t*)d_ws;
    half_t* Kh  = Qh  + QK;
    half_t* Vt  = Kh  + QK;
    half_t* Ctx = Vt  + QK;
    half_t* Wh  = Ctx + QK;                   // + H*H -> ~33 MB total f16

    // 1) pack / convert
    {
        int threads = 256;
        int blocks = (int)((QK + threads - 1) / threads);
        pack_qkv_kernel<<<blocks, threads, 0, stream>>>(Q, K, V, Qh, Kh, Vt);
        pack_w_kernel<<<(HH * HH + 255) / 256, 256, 0, stream>>>(W, Wh);
    }
    // 2) flash attention: B*NH*(S/64) blocks of 128 threads
    {
        int blocks = BB * NH * (SS / 64);     // 1024
        flash_attn_kernel<<<blocks, 128, 0, stream>>>(Qh, Kh, Vt, sg, lm, Ctx);
    }
    // 3) projection: (B*S/16)*(H/16) tiles, 8 waves per block
    {
        int tiles = (BB * SS / 16) * (HH / 16);   // 16384
        proj_kernel<<<tiles / 8, 256, 0, stream>>>(Ctx, Wh, bo, Out);
    }
}
